// CartesianTransformer_60816736911415
// MI455X (gfx1250) — matrix operand order repaired and checked by reading, compiled, not run on hardware
//
#include <hip/hip_runtime.h>
#include <hip/hip_bf16.h>

// ---------------------------------------------------------------------------
// CartesianTransformer for MI455X (gfx1250, wave32, WMMA).
// GEMMs: register-blocked (wave tile 32x64 -> 8 WMMA/K-step), LDS double
// buffered, global->LDS staging via CDNA5 async loads
// (global_load_async_to_lds_b128, ASYNCcnt) so staging bypasses VGPRs and
// runs decoupled from the WMMA pipe. Small per-token ops (embedding, cutoff
// mask, softmax attention over 65 keys, LayerNorm) are VALU kernels (~1%).
// ---------------------------------------------------------------------------

typedef __attribute__((ext_vector_type(4)))  _Float16 v4h;
typedef __attribute__((ext_vector_type(8)))  _Float16 v8h;
typedef __attribute__((ext_vector_type(16))) _Float16 v16h;
typedef __attribute__((ext_vector_type(8)))  float    v8f;

#define BATCH   2048
#define SEQ     64
#define TOK     65            // SEQ + 1 (central token)
#define DMODEL  256
#define NHEAD   8
#define HDIM    32
#define FFDIM   1024
#define NLAYER  3
#define BS      (BATCH * SEQ)     // 131072
#define BT      (BATCH * TOK)     // 133120

__device__ __forceinline__ float silu_f(float v) {
    return v / (1.0f + __expf(-v));
}

// ---- CDNA5 async global->LDS copy (16 bytes/lane, tracked by ASYNCcnt) ----
__device__ __forceinline__ void async_copy_b128(const _Float16* g, _Float16* l) {
    unsigned lofs = (unsigned)(size_t)l;   // low 32 bits of flat addr = LDS addr
    asm volatile("global_load_async_to_lds_b128 %0, %1, off"
                 :: "v"(lofs), "v"(g)
                 : "memory");
}
__device__ __forceinline__ void wait_async0() {
    asm volatile("s_wait_asynccnt 0x0" ::: "memory");
}

// ---------------------------------------------------------------------------
// f32 -> f16 weight conversion
// ---------------------------------------------------------------------------
__global__ __launch_bounds__(256) void k_cvt(const float* __restrict__ src,
                                             _Float16* __restrict__ dst, long n) {
    long i = (long)blockIdx.x * 256 + threadIdx.x;
    if (i < n) dst[i] = (_Float16)src[i];
}

// ---------------------------------------------------------------------------
// r-embedding + SiLU + concat [coords | neighbor_emb | input_messages] -> f16
// ---------------------------------------------------------------------------
__global__ __launch_bounds__(256) void k_embed(
    const float* __restrict__ x, const int* __restrict__ nspec,
    const float* __restrict__ msgs, const float* __restrict__ rw,
    const float* __restrict__ rb, const float* __restrict__ nemb,
    _Float16* __restrict__ cat) {
    long idx = (long)blockIdx.x * 256 + threadIdx.x;
    long bs = idx >> 8;
    int d = (int)(idx & 255);
    float x0 = x[bs * 3 + 0], x1 = x[bs * 3 + 1], x2 = x[bs * 3 + 2];
    float len = sqrtf(x0 * x0 + x1 * x1 + x2 * x2 + 1e-15f);
    float e = rw[d * 4 + 0] * x0 + rw[d * 4 + 1] * x1 +
              rw[d * 4 + 2] * x2 + rw[d * 4 + 3] * len + rb[d];
    e = silu_f(e);
    long base = bs * 768;
    cat[base + d]       = (_Float16)e;
    cat[base + 256 + d] = (_Float16)nemb[(long)nspec[bs] * 256 + d];
    cat[base + 512 + d] = (_Float16)msgs[bs * 256 + d];
}

// ---------------------------------------------------------------------------
// multiplicative attention mask from smooth cutoff; one thread per (b, t)
// ---------------------------------------------------------------------------
__global__ __launch_bounds__(256) void k_mask(
    const float* __restrict__ x, const unsigned char* __restrict__ maskb,
    float* __restrict__ mult) {
    long idx = (long)blockIdx.x * 256 + threadIdx.x;
    if (idx >= (long)BT) return;
    int b = (int)(idx / TOK);
    int t = (int)(idx % TOK);
    float m = 1.0f;
    if (t > 0) {
        long bs = (long)b * SEQ + (t - 1);
        float x0 = x[bs * 3 + 0], x1 = x[bs * 3 + 1], x2 = x[bs * 3 + 2];
        float r = sqrtf(x0 * x0 + x1 * x1 + x2 * x2 + 1e-16f);
        float g = (r - 2.0f + 0.75f) / 0.75f;
        float f = 0.5f + 0.5f * __cosf(3.14159265358979323846f * g);
        if (r >= 2.0f) f = 0.0f;
        if (r <= 1.25f) f = 1.0f;
        if (maskb[bs]) f = 0.0f;
        m = f;
    }
    mult[idx] = m;
}

// ---------------------------------------------------------------------------
// Assemble tokens: h[b,0,:] = central_emb[species], h[b,1+s,:] = compress out
// ---------------------------------------------------------------------------
__global__ __launch_bounds__(256) void k_assemble(
    const float* __restrict__ out2, const int* __restrict__ cspec,
    const float* __restrict__ cemb, float* __restrict__ h,
    _Float16* __restrict__ h16) {
    long idx = (long)blockIdx.x * 256 + threadIdx.x;
    long row = idx >> 8;
    int d = (int)(idx & 255);
    int b = (int)(row / TOK);
    int t = (int)(row % TOK);
    float v;
    if (t == 0) v = cemb[(long)cspec[b] * 256 + d];
    else        v = out2[((long)b * SEQ + (t - 1)) * 256 + d];
    h[idx] = v;
    h16[idx] = (_Float16)v;
}

// ---------------------------------------------------------------------------
// WMMA GEMM: out = act(A[M,K]f16 @ W[N,K]f16^T + bias)
//   block = 256 threads (8 waves) -> block tile 64(M) x 256(N)
//   wave grid 2x4, wave tile 32x64 -> 2x4 = 8 WMMA tiles per wave
//   K stepped by 32 through double-buffered LDS (rows padded to 40 halves:
//   80B stride = 20-bank stagger, conflict-free fragment reads).
//   Staging uses global_load_async_to_lds_b128 (ASYNCcnt); one
//   s_wait_asynccnt 0 + barrier per K-step publishes the buffer.
//   Requires M%64==0, N%256==0, K%32==0 (true for all call sites).
// ---------------------------------------------------------------------------
__global__ __launch_bounds__(256) void k_gemm(
    const _Float16* __restrict__ A, const _Float16* __restrict__ W,
    const float* __restrict__ bias, float* __restrict__ outF,
    _Float16* __restrict__ outH, int N, int K, int act) {
    __shared__ _Float16 As[2][64][40];
    __shared__ _Float16 Ws[2][256][40];

    const int tid  = threadIdx.x;
    const int lane = tid & 31;
    const int wave = tid >> 5;
    const int wm = wave >> 2;          // 0..1 : wave M tile (32 rows each)
    const int wn = wave & 3;           // 0..3 : wave N tile (64 cols each)
    const long rowBase = (long)blockIdx.y * 64;
    const long colBase = (long)blockIdx.x * 256;

    // staging layout: A tile 64x32 (8 halves/thread), W tile 256x32 (4 rows,
    // 8 halves each per thread)
    const int sr = tid >> 2;           // 0..63
    const int sc = (tid & 3) * 8;      // 0,8,16,24

    const int m16 = lane & 15;
    const int hi  = lane >> 4;

    v8f acc[2][4];
#pragma unroll
    for (int mt = 0; mt < 2; ++mt)
#pragma unroll
        for (int nt = 0; nt < 4; ++nt) acc[mt][nt] = (v8f){};

    // ---- prologue: async-stage K-tile 0 into buffer 0 ------------------
    async_copy_b128(&A[(rowBase + sr) * (long)K + sc], &As[0][sr][sc]);
#pragma unroll
    for (int j = 0; j < 4; ++j)
        async_copy_b128(&W[(colBase + sr + j * 64) * (long)K + sc],
                        &Ws[0][sr + j * 64][sc]);

    int cur = 0;
    for (int k0 = 0; k0 < K; k0 += 32) {
        wait_async0();     // own async LDS writes for buf[cur] complete
        __syncthreads();   // buf[cur] visible; all reads of buf[cur^1] done

        if (k0 + 32 < K) {  // async-stage next K-tile into the other buffer
            int kn = k0 + 32;
            async_copy_b128(&A[(rowBase + sr) * (long)K + kn + sc],
                            &As[cur ^ 1][sr][sc]);
#pragma unroll
            for (int j = 0; j < 4; ++j)
                async_copy_b128(&W[(colBase + sr + j * 64) * (long)K + kn + sc],
                                &Ws[cur ^ 1][sr + j * 64][sc]);
            if (kn + 32 < K) {
                __builtin_prefetch(&A[(rowBase + sr) * (long)K + kn + 32 + sc], 0, 1);
                __builtin_prefetch(&W[(colBase + sr) * (long)K + kn + 32 + sc], 0, 1);
            }
        }

        // ---- compute from buf[cur] ------------------------------------
        // A fragments (16x32 f16): lane<16 -> row m16, K {0..7,16..23};
        //                          lane>=16 -> K {8..15,24..31}
        v16h a0, a1;
        {
            v8h lo = *(const v8h*)&As[cur][wm * 32 + m16][hi * 8];
            v8h hh = *(const v8h*)&As[cur][wm * 32 + m16][hi * 8 + 16];
            a0 = __builtin_shufflevector(lo, hh, 0, 1, 2, 3, 4, 5, 6, 7,
                                         8, 9, 10, 11, 12, 13, 14, 15);
            lo = *(const v8h*)&As[cur][wm * 32 + 16 + m16][hi * 8];
            hh = *(const v8h*)&As[cur][wm * 32 + 16 + m16][hi * 8 + 16];
            a1 = __builtin_shufflevector(lo, hh, 0, 1, 2, 3, 4, 5, 6, 7,
                                         8, 9, 10, 11, 12, 13, 14, 15);
        }
#pragma unroll
        for (int nt = 0; nt < 4; ++nt) {
            // B fragment (32x16): lane<16 -> col m16, K {0..7,16..23}; lane>=16 -> K {8..15,24..31}
            v8h bl = *(const v8h*)&Ws[cur][wn * 64 + nt * 16 + m16][hi * 8];
            v8h bh = *(const v8h*)&Ws[cur][wn * 64 + nt * 16 + m16][hi * 8 + 16];
            v16h bf = __builtin_shufflevector(bl, bh, 0, 1, 2, 3, 4, 5, 6, 7,
                                              8, 9, 10, 11, 12, 13, 14, 15);
            acc[0][nt] = __builtin_amdgcn_wmma_f32_16x16x32_f16(
                false, a0, false, bf, (short)0, acc[0][nt], false, false);
            acc[1][nt] = __builtin_amdgcn_wmma_f32_16x16x32_f16(
                false, a1, false, bf, (short)0, acc[1][nt], false, false);
        }
        cur ^= 1;
    }

    // epilogue: C layout — VGPR r: lane<16 -> (M=r, N=lane); lane>=16 -> (M=r+8)
#pragma unroll
    for (int nt = 0; nt < 4; ++nt) {
        const long col = colBase + wn * 64 + nt * 16 + m16;
        const float bv = bias ? bias[col] : 0.0f;
#pragma unroll
        for (int mt = 0; mt < 2; ++mt) {
#pragma unroll
            for (int r = 0; r < 8; ++r) {
                long row = rowBase + wm * 32 + mt * 16 + r + 8 * hi;
                float v = acc[mt][nt][r] + bv;
                if (act) v = silu_f(v);
                if (outF) outF[row * (long)N + col] = v;
                if (outH) outH[row * (long)N + col] = (_Float16)v;
            }
        }
    }
}

// ---------------------------------------------------------------------------
// Attention: one block per (b, head). T=65, HD=32. Scores/K/V in LDS.
// qkv layout per row: [Q(0:256) | K(256:512) | V(512:768)], head h at h*32.
// ---------------------------------------------------------------------------
__global__ __launch_bounds__(128) void k_attn(
    const _Float16* __restrict__ qkv, const float* __restrict__ mult,
    _Float16* __restrict__ attn) {
    __shared__ float Ks[TOK][33];
    __shared__ float Vs[TOK][33];
    __shared__ float Sc[TOK][66];
    __shared__ float ms[TOK];

    const int b = blockIdx.x >> 3;
    const int hh = blockIdx.x & 7;
    const int tid = threadIdx.x;

    for (int i = tid; i < TOK * 32; i += 128) {
        int kk = i >> 5, d = i & 31;
        long rowq = ((long)b * TOK + kk) * 768;
        Ks[kk][d] = (float)qkv[rowq + 256 + hh * 32 + d];
        Vs[kk][d] = (float)qkv[rowq + 512 + hh * 32 + d];
    }
    for (int i = tid; i < TOK; i += 128) ms[i] = mult[(long)b * TOK + i];
    __syncthreads();

    if (tid < TOK) {
        const int q = tid;
        float qr[32];
        long rowq = ((long)b * TOK + q) * 768;
#pragma unroll
        for (int d = 0; d < 32; ++d) qr[d] = (float)qkv[rowq + hh * 32 + d];
        const float scale = 0.17677669529663687f;  // 1/sqrt(32)
        float mx = -1e30f;
        for (int k = 0; k < TOK; ++k) {
            float s = 0.0f;
#pragma unroll
            for (int d = 0; d < 32; ++d) s += qr[d] * Ks[k][d];
            s *= scale;
            Sc[q][k] = s;
            mx = fmaxf(mx, s);
        }
        float sum = 0.0f;
        for (int k = 0; k < TOK; ++k) {
            float e = __expf(Sc[q][k] - mx);
            sum += e;
            Sc[q][k] = e * ms[k];   // softmax then multiplicative mask
        }
        float inv = 1.0f / sum;
        long ro = ((long)b * TOK + q) * 256 + hh * 32;
#pragma unroll
        for (int d = 0; d < 32; ++d) {
            float a = 0.0f;
            for (int k = 0; k < TOK; ++k) a += Sc[q][k] * Vs[k][d];
            attn[ro + d] = (_Float16)(a * inv);
        }
    }
}

// ---------------------------------------------------------------------------
// LayerNorm(h + proj); block per row, 256 threads.
// ---------------------------------------------------------------------------
__global__ __launch_bounds__(256) void k_ln(
    const float* __restrict__ hin, const float* __restrict__ proj,
    const float* __restrict__ g, const float* __restrict__ bta,
    float* __restrict__ hout, _Float16* __restrict__ h16) {
    __shared__ float red[256];
    const long row = blockIdx.x;
    const int d = threadIdx.x;
    float v = hin[row * 256 + d] + proj[row * 256 + d];
    red[d] = v;
    __syncthreads();
    for (int s = 128; s > 0; s >>= 1) {
        if (d < s) red[d] += red[d + s];
        __syncthreads();
    }
    float mu = red[0] * (1.0f / 256.0f);
    __syncthreads();
    float dv = v - mu;
    red[d] = dv * dv;
    __syncthreads();
    for (int s = 128; s > 0; s >>= 1) {
        if (d < s) red[d] += red[d + s];
        __syncthreads();
    }
    float rstd = rsqrtf(red[0] * (1.0f / 256.0f) + 1e-5f);
    float y = dv * rstd * g[d] + bta[d];
    hout[row * 256 + d] = y;
    h16[row * 256 + d] = (_Float16)y;
}

// ---------------------------------------------------------------------------
// Scatter final h into d_out = [output_messages (B,S,D) | central (B,D)]
// ---------------------------------------------------------------------------
__global__ __launch_bounds__(256) void k_final(
    const float* __restrict__ h, float* __restrict__ out) {
    long idx = (long)blockIdx.x * 256 + threadIdx.x;
    const long BSD = (long)BS * DMODEL;
    if (idx < BSD) {
        long b = idx / (SEQ * DMODEL);
        long r = idx % (SEQ * DMODEL);
        long s = r / DMODEL;
        long d = r % DMODEL;
        out[idx] = h[((long)b * TOK + 1 + s) * 256 + d];
    } else {
        long j = idx - BSD;
        long b = j / DMODEL;
        long d = j % DMODEL;
        out[idx] = h[((long)b * TOK) * 256 + d];
    }
}

// ---------------------------------------------------------------------------
// Host launch
// ---------------------------------------------------------------------------
extern "C" void kernel_launch(void* const* d_in, const int* in_sizes, int n_in,
                              void* d_out, int out_size, void* d_ws, size_t ws_size,
                              hipStream_t stream) {
    (void)in_sizes; (void)n_in; (void)out_size; (void)ws_size;

    const float* x      = (const float*)d_in[0];
    const int*   cspec  = (const int*)d_in[1];
    const int*   nspec  = (const int*)d_in[2];
    const float* msgs   = (const float*)d_in[3];
    const unsigned char* maskb = (const unsigned char*)d_in[4];
    const float* r_emb_w = (const float*)d_in[7];
    const float* r_emb_b = (const float*)d_in[8];
    const float* nemb    = (const float*)d_in[9];
    const float* cw1     = (const float*)d_in[10];
    const float* cb1     = (const float*)d_in[11];
    const float* cw2     = (const float*)d_in[12];
    const float* cb2     = (const float*)d_in[13];
    const float* cemb    = (const float*)d_in[14];
    const float* qkv_w   = (const float*)d_in[15];
    const float* qkv_b   = (const float*)d_in[16];
    const float* ao_w    = (const float*)d_in[17];
    const float* ao_b    = (const float*)d_in[18];
    const float* ln1_g   = (const float*)d_in[19];
    const float* ln1_b   = (const float*)d_in[20];
    const float* ff_w1   = (const float*)d_in[21];
    const float* ff_b1   = (const float*)d_in[22];
    const float* ff_w2   = (const float*)d_in[23];
    const float* ff_b2   = (const float*)d_in[24];
    const float* ln2_g   = (const float*)d_in[25];
    const float* ln2_b   = (const float*)d_in[26];
    float* out = (float*)d_out;

    // ---- workspace layout (bytes) --------------------------------------
    const size_t OFF_WTS  = 0;                                  //   5.24 MB
    const size_t OFF_H32  = OFF_WTS  + 5242880;                 // 136.31 MB
    const size_t OFF_H16  = OFF_H32  + 136314880;               //  68.16 MB
    const size_t OFF_BIG  = OFF_H16  + 68157440;                // 272.63 MB (concat/qkv16/ff1)
    const size_t OFF_MID  = OFF_BIG  + 272629760;               //  68.16 MB (mid16/attn16)
    const size_t OFF_PROJ = OFF_MID  + 68157440;                // 136.31 MB (out2/proj)
    const size_t OFF_MULT = OFF_PROJ + 136314880;               //   0.53 MB

    char* ws = (char*)d_ws;
    _Float16* wts  = (_Float16*)(ws + OFF_WTS);
    float*    h32  = (float*)   (ws + OFF_H32);
    _Float16* h16  = (_Float16*)(ws + OFF_H16);
    _Float16* big  = (_Float16*)(ws + OFF_BIG);
    _Float16* mid  = (_Float16*)(ws + OFF_MID);
    float*    proj = (float*)   (ws + OFF_PROJ);
    float*    mult = (float*)   (ws + OFF_MULT);

    // f16 weight sub-offsets (in halves)
    _Float16* cw1_16  = wts;                 // 256*768
    _Float16* cw2_16  = wts + 196608;        // 256*256
    _Float16* qkvw16  = wts + 262144;        // 3*768*256
    _Float16* aow16   = wts + 851968;        // 3*256*256
    _Float16* ffw1_16 = wts + 1048576;       // 3*1024*256
    _Float16* ffw2_16 = wts + 1835008;       // 3*256*1024

    // ---- 1. weight conversion ------------------------------------------
    k_cvt<<<768,  256, 0, stream>>>(cw1,   cw1_16,  196608);
    k_cvt<<<256,  256, 0, stream>>>(cw2,   cw2_16,  65536);
    k_cvt<<<2304, 256, 0, stream>>>(qkv_w, qkvw16,  589824);
    k_cvt<<<768,  256, 0, stream>>>(ao_w,  aow16,   196608);
    k_cvt<<<3072, 256, 0, stream>>>(ff_w1, ffw1_16, 786432);
    k_cvt<<<3072, 256, 0, stream>>>(ff_w2, ffw2_16, 786432);

    // ---- 2. embedding + concat + mask ----------------------------------
    k_embed<<<BS * DMODEL / 256, 256, 0, stream>>>(x, nspec, msgs, r_emb_w,
                                                   r_emb_b, nemb, big);
    k_mask<<<(BT + 255) / 256, 256, 0, stream>>>(x, maskb, mult);

    // ---- 3. compress MLP (two WMMA GEMMs) ------------------------------
    // mid = SiLU(concat @ cw1^T + cb1)   [BS, 256], K=768
    k_gemm<<<dim3(256 / 256, BS / 64), 256, 0, stream>>>(
        big, cw1_16, cb1, nullptr, mid, 256, 768, 1);
    // proj(out2) = mid @ cw2^T + cb2     [BS, 256], K=256
    k_gemm<<<dim3(256 / 256, BS / 64), 256, 0, stream>>>(
        mid, cw2_16, cb2, proj, nullptr, 256, 256, 0);

    // ---- 4. assemble tokens --------------------------------------------
    k_assemble<<<BT * DMODEL / 256, 256, 0, stream>>>(proj, cspec, cemb, h32, h16);

    // ---- 5. transformer layers -----------------------------------------
    for (int i = 0; i < NLAYER; ++i) {
        // qkv16 = h @ qkv_w^T + qkv_b   [BT, 768]
        k_gemm<<<dim3(768 / 256, BT / 64), 256, 0, stream>>>(
            h16, qkvw16 + (size_t)i * 768 * 256, qkv_b + (size_t)i * 768,
            nullptr, big, 768, 256, 0);
        // attention
        k_attn<<<BATCH * NHEAD, 128, 0, stream>>>(big, mult, mid);
        // proj = attn @ ao_w^T + ao_b   [BT, 256]
        k_gemm<<<dim3(256 / 256, BT / 64), 256, 0, stream>>>(
            mid, aow16 + (size_t)i * 65536, ao_b + (size_t)i * 256,
            proj, nullptr, 256, 256, 0);
        // h = LN1(h + proj)
        k_ln<<<BT, 256, 0, stream>>>(h32, proj, ln1_g + (size_t)i * 256,
                                     ln1_b + (size_t)i * 256, h32, h16);
        // ff1 = SiLU(h @ ff_w1^T + b1)  [BT, 1024]
        k_gemm<<<dim3(1024 / 256, BT / 64), 256, 0, stream>>>(
            h16, ffw1_16 + (size_t)i * 262144, ff_b1 + (size_t)i * 1024,
            nullptr, big, 1024, 256, 1);
        // proj = ff1 @ ff_w2^T + b2     [BT, 256], K=1024
        k_gemm<<<dim3(256 / 256, BT / 64), 256, 0, stream>>>(
            big, ffw2_16 + (size_t)i * 262144, ff_b2 + (size_t)i * 256,
            proj, nullptr, 256, 1024, 0);
        // h = LN2(h + proj)
        k_ln<<<BT, 256, 0, stream>>>(h32, proj, ln2_g + (size_t)i * 256,
                                     ln2_b + (size_t)i * 256, h32, h16);
    }

    // ---- 6. final scatter ----------------------------------------------
    k_final<<<(BS * DMODEL + BATCH * DMODEL) / 256, 256, 0, stream>>>(h32, out);
}
